// Pillar_Transformer_87376814670476
// MI455X (gfx1250) — matrix operand order repaired and checked by reading, not compiled, not run on hardware
//
#include <hip/hip_runtime.h>
#include <hip/hip_bf16.h>
#include <math.h>

// ---------------- problem constants ----------------
#define B_     32
#define N_     2048
#define D_     768
#define T_     101
#define HEADS_ 12
#define HD_    64
#define S_     (B_*100)        // pillar cells
#define NP_    (B_*N_)         // 65536 points
#define MLPH_  3072
#define NC_    40
#define DEPTH_ 12
#define EPS_   1e-5f
#define SCALE_ 0.125f          // HD^-0.5
#define MTOK_  (B_*T_)         // 3232 token rows
#define MPAD_  3328            // 26*128: padded row count for GEMM A inputs

// ---------------- WMMA types (gfx1250, wave32) ----------------
typedef __attribute__((ext_vector_type(16))) __bf16 v16bf;
typedef __attribute__((ext_vector_type(8)))  __bf16 v8bf;
typedef __attribute__((ext_vector_type(8)))  float  v8f;

static __device__ __forceinline__ v16bf cat8(v8bf lo, v8bf hi) {
  return __builtin_shufflevector(lo, hi, 0,1,2,3,4,5,6,7,8,9,10,11,12,13,14,15);
}

// async global->LDS 16B copy (per-lane), tracked by ASYNCcnt
static __device__ __forceinline__ void async_g2l_b128(unsigned lds_off, const void* gaddr) {
  asm volatile("global_load_async_to_lds_b128 %0, %1, off"
               :: "v"(lds_off), "v"(gaddr) : "memory");
}
static __device__ __forceinline__ void wait_async0() {
  asm volatile("s_wait_asynccnt 0x0" ::: "memory");
}

// =====================================================================
// bf16-WMMA GEMM:  Out = epi( A[MxK]bf16 @ Bw[KxN]bf16 + bias )
// MODE 0: store bf16            MODE 1: bias + exact GELU -> bf16
// MODE 2: bias + residual -> f32 MODE 3: pillar (A = h||pooled[cell],
//                                        epi = relu -> atomicMax f32[cell])
// Block 256 thr = 8 waves; tile 128x64x64; wave -> 32x32 (2x2 wmma, 2 K-steps)
// A tile staged via GLOBAL_LOAD_ASYNC_TO_LDS_B128 (MODE!=3; buffers row-padded
// so no bounds branch: pad rows only feed discarded outputs).
// =====================================================================
#define BM 128
#define BN 64
#define BK 64
#define LDT 72   // padded LDS stride (elements); 144B rows, 16B aligned

template <int MODE>
__global__ __launch_bounds__(256)
void gemm_bf16_wmma(const __bf16* __restrict__ A, const __bf16* __restrict__ Bw,
                    const float* __restrict__ bias, const float* __restrict__ resid,
                    void* __restrict__ OutV, int M, int N, int K,
                    const __bf16* __restrict__ A2, const int* __restrict__ cellp)
{
  __shared__ __align__(16) __bf16 As[BM][LDT];
  __shared__ __align__(16) __bf16 Bs[BN][LDT];   // transposed: Bs[n][k]

  const int tid  = threadIdx.x;
  const int wave = tid >> 5;
  const int lane = tid & 31;
  const int wm   = wave >> 1;        // 0..3 -> 32-row group
  const int wn   = wave & 1;         // 0..1 -> 32-col group
  const int rowBase = blockIdx.y * BM;
  const int colBase = blockIdx.x * BN;
  const int hh  = lane >> 4;         // half-wave
  const int l16 = lane & 15;

  const int chunk = tid & 7;         // 16B chunk within 64-element row
  const int row0  = tid >> 3;        // 0..31

  v8f acc[2][2] = {};

  for (int kt = 0; kt < K; kt += BK) {
    // ---- stage A tile (128 x 64 bf16) ----
    if (MODE != 3) {
      unsigned asbase = (unsigned)(uintptr_t)(&As[0][0]);
      #pragma unroll
      for (int p = 0; p < 4; ++p) {
        int row = row0 + p * 32;
        const __bf16* g = A + (size_t)(rowBase + row) * K + kt + chunk * 8;
        async_g2l_b128(asbase + row * (LDT * 2) + chunk * 16, g);
      }
    } else {
      #pragma unroll
      for (int p = 0; p < 4; ++p) {
        int row  = row0 + p * 32;
        int grow = rowBase + row;
        int kk   = kt + chunk * 8;
        const __bf16* src = (kk < 128)
            ? (A  + (size_t)grow * 128 + kk)
            : (A2 + (size_t)cellp[grow] * 128 + (kk - 128));
        *(v8bf*)&As[row][chunk * 8] = *(const v8bf*)src;
      }
    }
    // ---- stage B tile transposed (global bf16 [K][N] -> LDS [n][k]) ----
    {
      int n8 = (tid & 7) * 8;
      int kr = tid >> 3;
      #pragma unroll
      for (int p = 0; p < 2; ++p) {
        int kk = kr + p * 32;
        v8bf v = *(const v8bf*)(Bw + (size_t)(kt + kk) * N + colBase + n8);
        #pragma unroll
        for (int q = 0; q < 8; ++q) Bs[n8 + q][kk] = v[q];
      }
    }
    if (kt + BK < K)  // pull next weight tile toward L2 (global_prefetch_b8)
      __builtin_prefetch(Bw + (size_t)(kt + BK) * N + colBase, 0, 1);
    if (MODE != 3) wait_async0();
    __syncthreads();

    // ---- 2 K-steps x (2x2) wmma per stage ----
    #pragma unroll
    for (int ks = 0; ks < BK; ks += 32) {
      v16bf af[2], bfr[2];
      #pragma unroll
      for (int i = 0; i < 2; ++i) {
        const __bf16* p = &As[wm * 32 + i * 16 + l16][ks];
        af[i] = cat8(*(const v8bf*)(p + 8 * hh), *(const v8bf*)(p + 16 + 8 * hh));
      }
      #pragma unroll
      for (int j = 0; j < 2; ++j) {
        const __bf16* p = &Bs[wn * 32 + j * 16 + l16][ks];
        bfr[j] = cat8(*(const v8bf*)(p + 8 * hh), *(const v8bf*)(p + 16 + 8 * hh));
      }
      #pragma unroll
      for (int i = 0; i < 2; ++i)
        #pragma unroll
        for (int j = 0; j < 2; ++j)
          acc[i][j] = __builtin_amdgcn_wmma_f32_16x16x32_bf16(
              false, af[i], false, bfr[j], (short)0, acc[i][j], false, false);
    }
    __syncthreads();
  }

  // ---- epilogue (C layout: VGPR e -> M = e + 8*hh, N = l16) ----
  #pragma unroll
  for (int i = 0; i < 2; ++i) {
    #pragma unroll
    for (int j = 0; j < 2; ++j) {
      int gcol = colBase + wn * 32 + j * 16 + l16;
      float bb = (MODE == 3 || bias == nullptr) ? 0.f : bias[gcol];
      #pragma unroll
      for (int e = 0; e < 8; ++e) {
        int grow = rowBase + wm * 32 + i * 16 + 8 * hh + e;
        if (grow >= M) continue;
        float v = acc[i][j][e] + bb;
        size_t idx = (size_t)grow * N + gcol;
        if (MODE == 0)      ((__bf16*)OutV)[idx] = (__bf16)v;
        else if (MODE == 1) ((__bf16*)OutV)[idx] =
                                (__bf16)(0.5f * v * (1.f + erff(v * 0.70710678f)));
        else if (MODE == 2) ((float*)OutV)[idx] = resid[idx] + v;
        else { // pillar: relu + segment-max scatter (non-negative => int cmp ok)
          int c = cellp[grow];
          atomicMax((int*)((float*)OutV + (size_t)c * N + gcol),
                    __float_as_int(fmaxf(v, 0.f)));
        }
      }
    }
  }
}

// =====================================================================
// fp32 -> bf16 conversion (weights once per GEMM; pooled once)
// =====================================================================
__global__ void cvt_bf16_kernel(const float* __restrict__ src, __bf16* __restrict__ dst, int n)
{
  int i = (blockIdx.x * blockDim.x + threadIdx.x) * 4;
  if (i >= n) return;
  float4 v = *(const float4*)(src + i);
  dst[i]     = (__bf16)v.x;
  dst[i + 1] = (__bf16)v.y;
  dst[i + 2] = (__bf16)v.z;
  dst[i + 3] = (__bf16)v.w;
}

// =====================================================================
// Pillar stage
// =====================================================================
__global__ void point_cell_kernel(const float* __restrict__ x, int* __restrict__ cell,
                                  float* __restrict__ pbuf, float* __restrict__ cnt,
                                  float* __restrict__ cen)
{
  int i = blockIdx.x * blockDim.x + threadIdx.x;
  if (i >= NP_) return;
  float x0 = x[(size_t)i*3], x1 = x[(size_t)i*3+1], x2 = x[(size_t)i*3+2];
  float p0 = x1, p1 = x0, p2 = x2;                 // permute [1,0,2]
  float y0 = fminf(fmaxf(p1 + 1.f, 0.f), 1.99f);
  float y1 = fminf(fmaxf(p2 + 1.f, 0.f), 1.99f);
  int i0 = (int)floorf(y0 * 5.f);
  int i1 = (int)floorf(y1 * 5.f);
  int b = i / N_;
  int c = i0 * 10 + i1 + b * 100;
  cell[i] = c;
  pbuf[(size_t)i*3] = p0; pbuf[(size_t)i*3+1] = p1; pbuf[(size_t)i*3+2] = p2;
  atomicAdd(&cnt[c], 1.f);
  atomicAdd(&cen[c*3+0], p0);
  atomicAdd(&cen[c*3+1], p1);
  atomicAdd(&cen[c*3+2], p2);
}

__global__ void centroid_div_kernel(const float* __restrict__ cnt, float* __restrict__ cen)
{
  int c = blockIdx.x * blockDim.x + threadIdx.x;
  if (c >= S_) return;
  float n = 1.f / fmaxf(cnt[c], 1.f);
  cen[c*3+0] *= n; cen[c*3+1] *= n; cen[c*3+2] *= n;
}

// per-point 6->32->64->128 MLP, weights in LDS; h -> bf16, segment_max -> f32
__global__ __launch_bounds__(256)
void point_mlp_kernel(const float* __restrict__ pbuf, const int* __restrict__ cell,
                      const float* __restrict__ cen,
                      const float* __restrict__ W1, const float* __restrict__ W2,
                      const float* __restrict__ W3,
                      __bf16* __restrict__ hbuf, float* __restrict__ pooled)
{
  __shared__ float w1[6*32], w2[32*64], w3[64*128];
  int tid = threadIdx.x;
  for (int i = tid; i < 6*32;   i += 256) w1[i] = W1[i];
  for (int i = tid; i < 32*64;  i += 256) w2[i] = W2[i];
  for (int i = tid; i < 64*128; i += 256) w3[i] = W3[i];
  __syncthreads();
  int i = blockIdx.x * 256 + tid;
  if (i >= NP_) return;
  int c = cell[i];
  float aug[6];
  aug[0] = pbuf[(size_t)i*3];   aug[1] = pbuf[(size_t)i*3+1]; aug[2] = pbuf[(size_t)i*3+2];
  aug[3] = aug[0] - cen[c*3];   aug[4] = aug[1] - cen[c*3+1]; aug[5] = aug[2] - cen[c*3+2];
  float h1[32];
  #pragma unroll
  for (int j = 0; j < 32; ++j) {
    float a = 0.f;
    #pragma unroll
    for (int k = 0; k < 6; ++k) a += aug[k] * w1[k*32+j];
    h1[j] = fmaxf(a, 0.f);
  }
  float h2[64];
  #pragma unroll
  for (int j = 0; j < 64; ++j) {
    float a = 0.f;
    #pragma unroll
    for (int k = 0; k < 32; ++k) a += h1[k] * w2[k*64+j];
    h2[j] = fmaxf(a, 0.f);
  }
  for (int d = 0; d < 128; ++d) {
    float a = 0.f;
    #pragma unroll
    for (int k = 0; k < 64; ++k) a += h2[k] * w3[k*128+d];
    a = fmaxf(a, 0.f);
    hbuf[(size_t)i*128 + d] = (__bf16)a;
    atomicMax((int*)&pooled[(size_t)c*128 + d], __float_as_int(a));
  }
}

// BN + cls + pos_embed -> tokens[B][T][D] (f32 residual stream)
__global__ void build_tokens_kernel(const float* __restrict__ pillar, const float* __restrict__ cls,
                                    const float* __restrict__ pos,
                                    const float* __restrict__ bn_g, const float* __restrict__ bn_b,
                                    const float* __restrict__ bn_mean, const float* __restrict__ bn_var,
                                    float* __restrict__ tokens)
{
  int idx = blockIdx.x * blockDim.x + threadIdx.x;
  if (idx >= B_ * T_ * D_) return;
  int d = idx % D_;
  int t = (idx / D_) % T_;
  int b = idx / (T_ * D_);
  float v;
  if (t == 0) v = cls[d];
  else {
    int s = t - 1;
    float inv = bn_g[s] * rsqrtf(bn_var[s] + EPS_);
    v = (pillar[((size_t)(b*100 + s)) * D_ + d] - bn_mean[s]) * inv + bn_b[s];
  }
  tokens[idx] = v + pos[t * D_ + d];
}

// =====================================================================
// Transformer helpers
// =====================================================================
template <typename OutT>
__global__ __launch_bounds__(256)
void ln_kernel(const float* __restrict__ in, OutT* __restrict__ out,
               const float* __restrict__ g, const float* __restrict__ b,
               int inRowStride, int outRowStride)
{
  __shared__ float red[256];
  int r = blockIdx.x, tid = threadIdx.x;
  const float* x = in  + (size_t)r * inRowStride;
  OutT*        y = out + (size_t)r * outRowStride;
  float v0 = x[tid], v1 = x[tid + 256], v2 = x[tid + 512];
  red[tid] = v0 + v1 + v2; __syncthreads();
  for (int o = 128; o > 0; o >>= 1) { if (tid < o) red[tid] += red[tid + o]; __syncthreads(); }
  float mean = red[0] * (1.f / 768.f);
  __syncthreads();
  float d0 = v0 - mean, d1 = v1 - mean, d2 = v2 - mean;
  red[tid] = d0*d0 + d1*d1 + d2*d2; __syncthreads();
  for (int o = 128; o > 0; o >>= 1) { if (tid < o) red[tid] += red[tid + o]; __syncthreads(); }
  float inv = rsqrtf(red[0] * (1.f / 768.f) + EPS_);
  y[tid]       = (OutT)(d0 * inv * g[tid]       + b[tid]);
  y[tid + 256] = (OutT)(d1 * inv * g[tid + 256] + b[tid + 256]);
  y[tid + 512] = (OutT)(d2 * inv * g[tid + 512] + b[tid + 512]);
}

// one block per (b, head, query): scores -> softmax -> @V  (<2.5% of FLOPs)
__global__ __launch_bounds__(128)
void attention_kernel(const __bf16* __restrict__ qkv, __bf16* __restrict__ attnout)
{
  __shared__ float qv[HD_];
  __shared__ float sc[128];
  __shared__ float red[128];
  int blk = blockIdx.x;
  int q  = blk % T_;
  int bh = blk / T_;
  int h  = bh % HEADS_;
  int b  = bh / HEADS_;
  int tid = threadIdx.x;
  const size_t rowQ = ((size_t)(b*T_ + q)) * (3*D_) + h * HD_;
  if (tid < HD_) qv[tid] = (float)qkv[rowQ + tid];
  __syncthreads();
  float s = -1e30f;
  if (tid < T_) {
    const __bf16* kr = qkv + ((size_t)(b*T_ + tid)) * (3*D_) + D_ + h * HD_;
    float a = 0.f;
    #pragma unroll
    for (int d = 0; d < HD_; ++d) a += qv[d] * (float)kr[d];
    s = a * SCALE_;
  }
  sc[tid] = s; red[tid] = s; __syncthreads();
  for (int o = 64; o > 0; o >>= 1) { if (tid < o) red[tid] = fmaxf(red[tid], red[tid + o]); __syncthreads(); }
  float mx = red[0]; __syncthreads();
  float e = (tid < T_) ? __expf(sc[tid] - mx) : 0.f;
  sc[tid] = e; red[tid] = e; __syncthreads();
  for (int o = 64; o > 0; o >>= 1) { if (tid < o) red[tid] += red[tid + o]; __syncthreads(); }
  float inv = 1.f / red[0]; __syncthreads();
  if (tid < HD_) {
    float o = 0.f;
    for (int t = 0; t < T_; ++t)
      o += sc[t] * (float)qkv[((size_t)(b*T_ + t)) * (3*D_) + 2*D_ + h*HD_ + tid];
    attnout[((size_t)(b*T_ + q)) * D_ + h*HD_ + tid] = (__bf16)(o * inv);
  }
}

__global__ __launch_bounds__(64)
void head_kernel(const float* __restrict__ clsln, const float* __restrict__ hw,
                 const float* __restrict__ hb, float* __restrict__ out)
{
  int b = blockIdx.x, c = threadIdx.x;
  if (c >= NC_) return;
  float a = hb[c];
  for (int d = 0; d < D_; ++d) a += clsln[(size_t)b*D_ + d] * hw[d*NC_ + c];
  out[b*NC_ + c] = a;
}

// =====================================================================
// host-side orchestration
// =====================================================================
static inline int cdiv(int a, int b) { return (a + b - 1) / b; }

extern "C" void kernel_launch(void* const* d_in, const int* in_sizes, int n_in,
                              void* d_out, int out_size, void* d_ws, size_t ws_size,
                              hipStream_t stream)
{
  (void)in_sizes; (void)n_in; (void)out_size; (void)ws_size;
  const float* x       = (const float*)d_in[0];
  const float* W1      = (const float*)d_in[1];
  const float* W2      = (const float*)d_in[2];
  const float* W3      = (const float*)d_in[3];
  const float* Wa      = (const float*)d_in[4];
  const float* bn_g    = (const float*)d_in[5];
  const float* bn_b    = (const float*)d_in[6];
  const float* bn_mean = (const float*)d_in[7];
  const float* bn_var  = (const float*)d_in[8];
  const float* cls_tok = (const float*)d_in[9];
  const float* pos     = (const float*)d_in[10];
  const float* ln1_g   = (const float*)d_in[11];
  const float* ln1_b   = (const float*)d_in[12];
  const float* qkv_w   = (const float*)d_in[13];
  const float* proj_w  = (const float*)d_in[14];
  const float* proj_b  = (const float*)d_in[15];
  const float* ln2_g   = (const float*)d_in[16];
  const float* ln2_b   = (const float*)d_in[17];
  const float* fc1_w   = (const float*)d_in[18];
  const float* fc1_b   = (const float*)d_in[19];
  const float* fc2_w   = (const float*)d_in[20];
  const float* fc2_b   = (const float*)d_in[21];
  const float* norm_g  = (const float*)d_in[22];
  const float* norm_b  = (const float*)d_in[23];
  const float* head_w  = (const float*)d_in[24];
  const float* head_b  = (const float*)d_in[25];
  float* out = (float*)d_out;

  // ---- workspace carve (256B aligned) ----
  char* ws = (char*)d_ws;
  size_t off = 0;
  auto carve = [&](size_t bytes) -> void* {
    void* p = ws + off;
    off += (bytes + 255) & ~(size_t)255;
    return p;
  };
  int*    cell      = (int*)   carve((size_t)NP_ * 4);
  float*  pbuf      = (float*) carve((size_t)NP_ * 3 * 4);
  float*  cnt       = (float*) carve((size_t)S_ * 4);        // -- zero region
  float*  centroid  = (float*) carve((size_t)S_ * 3 * 4);
  float*  pooled    = (float*) carve((size_t)S_ * 128 * 4);
  float*  pillar    = (float*) carve((size_t)S_ * D_ * 4);   // -- zero region end
  char*   zend      = ws + off;
  __bf16* hbuf      = (__bf16*)carve((size_t)NP_ * 128 * 2);
  __bf16* pooled_bf = (__bf16*)carve((size_t)S_ * 128 * 2);
  float*  tokens    = (float*) carve((size_t)MTOK_ * D_ * 4);
  __bf16* ybuf      = (__bf16*)carve((size_t)MPAD_ * D_ * 2);       // padded
  __bf16* qkvbuf    = (__bf16*)carve((size_t)MPAD_ * 3 * D_ * 2);   // padded
  __bf16* attnout   = (__bf16*)carve((size_t)MPAD_ * D_ * 2);       // padded
  __bf16* mlph      = (__bf16*)carve((size_t)MPAD_ * MLPH_ * 2);    // padded
  __bf16* wbf       = (__bf16*)carve((size_t)D_ * MLPH_ * 2);       // weight scratch
  __bf16* wabf      = (__bf16*)carve((size_t)256 * D_ * 2);
  float*  clsln     = (float*) carve((size_t)B_ * D_ * 4);

  // zero atomic-accumulated buffers (capture-safe async memset)
  hipMemsetAsync(cnt, 0, (size_t)(zend - (char*)cnt), stream);

  // ---- pillar stage ----
  point_cell_kernel<<<cdiv(NP_, 256), 256, 0, stream>>>(x, cell, pbuf, cnt, centroid);
  centroid_div_kernel<<<cdiv(S_, 256), 256, 0, stream>>>(cnt, centroid);
  point_mlp_kernel<<<NP_ / 256, 256, 0, stream>>>(pbuf, cell, centroid, W1, W2, W3, hbuf, pooled);
  cvt_bf16_kernel<<<cdiv(S_ * 128, 1024), 256, 0, stream>>>(pooled, pooled_bf, S_ * 128);
  cvt_bf16_kernel<<<cdiv(256 * D_, 1024), 256, 0, stream>>>(Wa, wabf, 256 * D_);
  // h2 = relu([h, pooled[cell]] @ Wa) fused with segment_max scatter into pillar
  gemm_bf16_wmma<3><<<dim3(D_ / BN, NP_ / BM), 256, 0, stream>>>(
      hbuf, wabf, nullptr, nullptr, pillar, NP_, D_, 256, pooled_bf, cell);
  build_tokens_kernel<<<cdiv(B_ * T_ * D_, 256), 256, 0, stream>>>(
      pillar, cls_tok, pos, bn_g, bn_b, bn_mean, bn_var, tokens);

  // ---- transformer ----
  const dim3 gQKV(3 * D_ / BN, MPAD_ / BM);
  const dim3 gD  (D_ / BN,     MPAD_ / BM);
  const dim3 gH  (MLPH_ / BN,  MPAD_ / BM);
  for (int l = 0; l < DEPTH_; ++l) {
    ln_kernel<__bf16><<<MTOK_, 256, 0, stream>>>(tokens, ybuf, ln1_g + l*D_, ln1_b + l*D_, D_, D_);
    cvt_bf16_kernel<<<cdiv(D_ * 3 * D_, 1024), 256, 0, stream>>>(
        qkv_w + (size_t)l * D_ * 3 * D_, wbf, D_ * 3 * D_);
    gemm_bf16_wmma<0><<<gQKV, 256, 0, stream>>>(
        ybuf, wbf, nullptr, nullptr, qkvbuf, MTOK_, 3 * D_, D_, nullptr, nullptr);
    attention_kernel<<<B_ * HEADS_ * T_, 128, 0, stream>>>(qkvbuf, attnout);
    cvt_bf16_kernel<<<cdiv(D_ * D_, 1024), 256, 0, stream>>>(
        proj_w + (size_t)l * D_ * D_, wbf, D_ * D_);
    gemm_bf16_wmma<2><<<gD, 256, 0, stream>>>(
        attnout, wbf, proj_b + l*D_, tokens, tokens, MTOK_, D_, D_, nullptr, nullptr);
    ln_kernel<__bf16><<<MTOK_, 256, 0, stream>>>(tokens, ybuf, ln2_g + l*D_, ln2_b + l*D_, D_, D_);
    cvt_bf16_kernel<<<cdiv(D_ * MLPH_, 1024), 256, 0, stream>>>(
        fc1_w + (size_t)l * D_ * MLPH_, wbf, D_ * MLPH_);
    gemm_bf16_wmma<1><<<gH, 256, 0, stream>>>(
        ybuf, wbf, fc1_b + l*MLPH_, nullptr, mlph, MTOK_, MLPH_, D_, nullptr, nullptr);
    cvt_bf16_kernel<<<cdiv(MLPH_ * D_, 1024), 256, 0, stream>>>(
        fc2_w + (size_t)l * MLPH_ * D_, wbf, MLPH_ * D_);
    gemm_bf16_wmma<2><<<gD, 256, 0, stream>>>(
        mlph, wbf, fc2_b + l*D_, tokens, tokens, MTOK_, D_, MLPH_, nullptr, nullptr);
  }

  // ---- final LN on cls rows + head ----
  ln_kernel<float><<<B_, 256, 0, stream>>>(tokens, clsln, norm_g, norm_b, T_ * D_, D_);
  head_kernel<<<B_, 64, 0, stream>>>(clsln, head_w, head_b, out);
}
